// IsoStreamMixing_257698038385
// MI455X (gfx1250) — matrix operand order repaired and checked by reading, not compiled, not run on hardware
//
#include <hip/hip_runtime.h>
#include <cmath>
#include <cstdint>

typedef __attribute__((ext_vector_type(16))) __bf16 v16bf;
typedef __attribute__((ext_vector_type(8)))  __bf16 v8bf;
typedef __attribute__((ext_vector_type(8)))  float  v8f;
typedef __attribute__((ext_vector_type(4)))  int    v4i;

constexpr int NDIM = 2048;
constexpr int BM = 128, BN = 128, BK = 64;
constexpr int LDS_STRIDE = BK + 8;   // 72 bf16 = 144 B: 16B-aligned rows, bank-spread

// ---------- gfx1250 async global->LDS path (guarded; falls back to b128 ld/st) ----------
#if defined(__HIP_DEVICE_COMPILE__) && __has_builtin(__builtin_amdgcn_global_load_async_to_lds_b128) && __has_builtin(__builtin_amdgcn_s_wait_asynccnt)
#define ISO_ASYNC_LDS 1
#else
#define ISO_ASYNC_LDS 0
#endif

#if ISO_ASYNC_LDS
__device__ __forceinline__ void iso_async_b128(const void* g, void* l) {
  __builtin_amdgcn_global_load_async_to_lds_b128(
      (__attribute__((address_space(1))) v4i*)(uintptr_t)g,
      (__attribute__((address_space(3))) v4i*)l,
      /*offset=*/0, /*cpol=*/0);
}
#endif

// ---------------- O(n^2) helper kernels (Householder rank-1 machinery) ----------------

template<typename T>
__global__ __launch_bounds__(256)
void iso_rowdot_w(const T* __restrict__ in, float* __restrict__ t1, float w0, float w1) {
  __shared__ float sm[256];
  const int r = blockIdx.x;
  float s = 0.f;
  for (int c = threadIdx.x; c < NDIM; c += 256) s += (float)in[(size_t)r * NDIM + c];
  sm[threadIdx.x] = s;
  __syncthreads();
  for (int o = 128; o > 0; o >>= 1) {
    if (threadIdx.x < o) sm[threadIdx.x] += sm[threadIdx.x + o];
    __syncthreads();
  }
  if (threadIdx.x == 0) t1[r] = w1 * sm[0] + (w0 - w1) * (float)in[(size_t)r * NDIM];
}

template<typename T>
__global__ __launch_bounds__(256)
void iso_coldot_w(const T* __restrict__ in, float* __restrict__ t2, float w0, float w1) {
  const int c = blockIdx.x * 256 + threadIdx.x;
  float s = 0.f;
  for (int r = 0; r < NDIM; ++r) s += (float)in[(size_t)r * NDIM + c];
  t2[c] = w1 * s + (w0 - w1) * (float)in[c];
}

__global__ __launch_bounds__(256)
void iso_wdot_scalar(const float* __restrict__ t1, float* __restrict__ s_out, float w0, float w1) {
  __shared__ float sm[256];
  float s = 0.f;
  for (int i = threadIdx.x; i < NDIM; i += 256) s += t1[i];
  sm[threadIdx.x] = s;
  __syncthreads();
  for (int o = 128; o > 0; o >>= 1) {
    if (threadIdx.x < o) sm[threadIdx.x] += sm[threadIdx.x + o];
    __syncthreads();
  }
  if (threadIdx.x == 0) s_out[0] = w1 * sm[0] + (w0 - w1) * t1[0];
}

// out = addC + (P in P), optional zero border + per-row sum-of-squares partials
template<typename T, bool ZERO_BORDER, bool PARTIALS>
__global__ __launch_bounds__(256)
void iso_combine_rank1(const T* __restrict__ in, float* __restrict__ out,
                       const float* __restrict__ t1, const float* __restrict__ t2,
                       const float* __restrict__ s_ptr, float addC,
                       float* __restrict__ partials, float w0, float w1) {
  __shared__ float sm[256];
  const int r = blockIdx.x;
  const float wr  = (r == 0) ? w0 : w1;
  const float s   = s_ptr[0];
  const float t1r = t1[r];
  float local = 0.f;
  for (int c = threadIdx.x; c < NDIM; c += 256) {
    const float wc = (c == 0) ? w0 : w1;
    float v = addC + (float)in[(size_t)r * NDIM + c]
              - 2.f * wr * t2[c] - 2.f * t1r * wc + 4.f * s * wr * wc;
    if (ZERO_BORDER && (r == 0 || c == 0)) v = 0.f;
    out[(size_t)r * NDIM + c] = v;
    if (PARTIALS) local += v * v;
  }
  if (PARTIALS) {
    sm[threadIdx.x] = local;
    __syncthreads();
    for (int o = 128; o > 0; o >>= 1) {
      if (threadIdx.x < o) sm[threadIdx.x] += sm[threadIdx.x + o];
      __syncthreads();
    }
    if (threadIdx.x == 0) partials[r] = sm[0];
  }
}

__global__ __launch_bounds__(256)
void iso_reduce_invnorm(const float* __restrict__ partials, float* __restrict__ invnorm) {
  __shared__ float sm[256];
  float s = 0.f;
  for (int i = threadIdx.x; i < NDIM; i += 256) s += partials[i];
  sm[threadIdx.x] = s;
  __syncthreads();
  for (int o = 128; o > 0; o >>= 1) {
    if (threadIdx.x < o) sm[threadIdx.x] += sm[threadIdx.x + o];
    __syncthreads();
  }
  if (threadIdx.x == 0) invnorm[0] = 1.f / sqrtf(sm[0]);
}

__global__ __launch_bounds__(256)
void iso_scale_to_bf16(const float* __restrict__ x0, const float* __restrict__ invnorm,
                       __bf16* __restrict__ xb) {
  const int i = blockIdx.x * 256 + threadIdx.x;
  xb[i] = (__bf16)(x0[i] * invnorm[0]);
}

// ---------------- WMMA bf16 GEMM engine ----------------
// MODE 0:  C = A * B^T           (Z = X X^T; pass Ain = Bin = X)  -- all staging straight copies
// MODE 1:  C = 1.5*B - 0.5*A*B   (X' = 1.5 X - 0.5 Z X; Ain = Z, Bin = X)

// straight row-major tile copy: BM rows x BK bf16 into lds[.][LDS_STRIDE]
__device__ __forceinline__ void iso_stage_tile(const __bf16* __restrict__ g,
                                               __bf16 (*lds)[LDS_STRIDE], int tid) {
#pragma unroll
  for (int i = 0; i < (BM * BK / 8) / 256; ++i) {   // 1024 16B-chunks / 256 thr = 4 each
    const int q   = tid + i * 256;
    const int row = q >> 3;
    const int cx  = (q & 7) * 8;
#if ISO_ASYNC_LDS
    iso_async_b128(g + (size_t)row * NDIM + cx, &lds[row][cx]);
#else
    *(v8bf*)&lds[row][cx] = *(const v8bf*)(g + (size_t)row * NDIM + cx);
#endif
  }
}

template<int MODE>
__global__ __launch_bounds__(256)
void iso_gemm_ns(const __bf16* __restrict__ Ain, const __bf16* __restrict__ Bin,
                 __bf16* __restrict__ Cout) {
  __shared__ __align__(16) __bf16 As[BM][LDS_STRIDE];  // A tile, row-major
  __shared__ __align__(16) __bf16 Bt[BN][LDS_STRIDE];  // B tile, stored as Bt[col][k]

  const int tid  = threadIdx.x;
  const int lane = tid & 31;
  const int wave = tid >> 5;
  const int waveM = (wave >> 2) * 64;  // 2 waves along M
  const int waveN = (wave & 3) * 32;   // 4 waves along N
  const int rowBase = blockIdx.y * BM;
  const int colBase = blockIdx.x * BN;
  const int hf = lane >> 4;
  const int ln = lane & 15;

  const v8f vzero = {0.f, 0.f, 0.f, 0.f, 0.f, 0.f, 0.f, 0.f};
  v8f acc[4][2];
#pragma unroll
  for (int i = 0; i < 4; ++i)
#pragma unroll
    for (int j = 0; j < 2; ++j) acc[i][j] = vzero;

  for (int k0 = 0; k0 < NDIM; k0 += BK) {
    // ---- stage A (always a straight copy) ----
    iso_stage_tile(Ain + (size_t)rowBase * NDIM + k0, As, tid);

    if (MODE == 0) {
      // B = X^T  =>  Bt[c][k] = X[colBase+c][k0+k] : straight copy too
      iso_stage_tile(Bin + (size_t)colBase * NDIM + k0, Bt, tid);
    } else {
      // B = X    =>  Bt[c][k] = X[(k0+k)][colBase+c] : transposing fill (coalesced reads)
      if (k0 + BK < NDIM)
        __builtin_prefetch(&Bin[(size_t)(k0 + BK + (tid >> 7)) * NDIM + colBase + (tid & 127)], 0, 0);
#pragma unroll 4
      for (int i = tid; i < BN * BK; i += 256) {
        const int k = i >> 7;      // 0..63
        const int c = i & 127;     // contiguous across threads -> coalesced global reads
        Bt[c][k] = Bin[(size_t)(k0 + k) * NDIM + colBase + c];
      }
    }
#if ISO_ASYNC_LDS
    __builtin_amdgcn_s_wait_asynccnt(0);
#endif
    __syncthreads();

    // ---- two K=32 WMMA steps per stage ----
#pragma unroll
    for (int k32 = 0; k32 < BK; k32 += 32) {
      const int kb = k32 >> 3;  // v8bf chunk base within a row (0 or 4)

      // A 16x32 fragment: lane<16 -> {K=k32+0..7, K=k32+16..23}; lane>=16 -> +8
      v16bf af[4];
#pragma unroll
      for (int i = 0; i < 4; ++i) {
        const v8bf* ar = (const v8bf*)&As[waveM + i * 16 + ln][0];
        af[i] = __builtin_shufflevector(ar[kb + hf], ar[kb + 2 + hf],
                                        0,1,2,3,4,5,6,7,8,9,10,11,12,13,14,15);
      }
      // B 32x16 fragment: lane holds col, lane<16 -> {K=k32+0..7, K=k32+16..23}; lane>=16 -> +8
      v16bf bfr[2];
#pragma unroll
      for (int j = 0; j < 2; ++j) {
        const v8bf* br = (const v8bf*)&Bt[waveN + j * 16 + ln][0];
        bfr[j] = __builtin_shufflevector(br[kb + hf], br[kb + 2 + hf],
                                         0,1,2,3,4,5,6,7,8,9,10,11,12,13,14,15);
      }

#pragma unroll
      for (int i = 0; i < 4; ++i)
#pragma unroll
        for (int j = 0; j < 2; ++j)
          acc[i][j] = __builtin_amdgcn_wmma_f32_16x16x32_bf16(
              false, af[i], false, bfr[j], (short)0, acc[i][j], false, false);
    }
    __syncthreads();
  }

  // ---- epilogue: D layout: VGPR v -> row v + 8*hf, col = ln ----
#pragma unroll
  for (int i = 0; i < 4; ++i) {
#pragma unroll
    for (int j = 0; j < 2; ++j) {
#pragma unroll
      for (int v = 0; v < 8; ++v) {
        const int row = rowBase + waveM + i * 16 + v + hf * 8;
        const int col = colBase + waveN + j * 16 + ln;
        float val = acc[i][j][v];
        if (MODE == 1) {
          const float x = (float)Bin[(size_t)row * NDIM + col];  // Bin = X
          val = 1.5f * x - 0.5f * val;
        }
        Cout[(size_t)row * NDIM + col] = (__bf16)val;
      }
    }
  }
}

// ---------------- host orchestration ----------------
extern "C" void kernel_launch(void* const* d_in, const int* in_sizes, int n_in,
                              void* d_out, int out_size, void* d_ws, size_t ws_size,
                              hipStream_t stream) {
  (void)in_sizes; (void)n_in; (void)out_size; (void)ws_size;
  const float* Hraw = (const float*)d_in[0];   // [2048,2048] f32
  float* Xf = (float*)d_out;                   // f32 scratch, overwritten by final H

  const size_t MAT_BF16 = (size_t)NDIM * NDIM * sizeof(__bf16);  // 8 MiB
  char* wsb = (char*)d_ws;
  __bf16* Xa = (__bf16*)(wsb);
  __bf16* Xb = (__bf16*)(wsb + MAT_BF16);
  __bf16* Zb = (__bf16*)(wsb + 2 * MAT_BF16);
  float*  t1 = (float*)(wsb + 3 * MAT_BF16);
  float*  t2 = t1 + NDIM;
  float*  sscal   = t2 + NDIM;
  float*  invnorm = sscal + 8;
  float*  parts   = invnorm + 8;

  // Householder vector: w = (ones/sqrt(n) - e1) normalized
  const double cc = 1.0 / sqrt((double)NDIM);
  const double wn = sqrt(2.0 - 2.0 * cc);
  const float w0 = (float)((cc - 1.0) / wn);
  const float w1 = (float)(cc / wn);

  const dim3 blk(256);
  const dim3 gRow(NDIM);
  const dim3 gCol(NDIM / 256);
  const dim3 gOne(1);
  const dim3 gElem((NDIM * NDIM) / 256);
  const dim3 gGemm(NDIM / BN, NDIM / BM);

  // --- A = (P Hraw P) embedded with zero border, and its Frobenius norm ---
  iso_rowdot_w<float><<<gRow, blk, 0, stream>>>(Hraw, t1, w0, w1);
  iso_coldot_w<float><<<gCol, blk, 0, stream>>>(Hraw, t2, w0, w1);
  iso_wdot_scalar<<<gOne, blk, 0, stream>>>(t1, sscal, w0, w1);
  iso_combine_rank1<float, true, true><<<gRow, blk, 0, stream>>>(
      Hraw, Xf, t1, t2, sscal, 0.f, parts, w0, w1);
  iso_reduce_invnorm<<<gOne, blk, 0, stream>>>(parts, invnorm);
  iso_scale_to_bf16<<<gElem, blk, 0, stream>>>(Xf, invnorm, Xa);

  // --- 5 Newton-Schulz steps: Z = X X^T ; X = 1.5 X - 0.5 Z X ---
  __bf16* cur = Xa;
  __bf16* nxt = Xb;
  for (int it = 0; it < 5; ++it) {
    iso_gemm_ns<0><<<gGemm, blk, 0, stream>>>(cur, cur, Zb);
    iso_gemm_ns<1><<<gGemm, blk, 0, stream>>>(Zb, cur, nxt);
    __bf16* tmp = cur; cur = nxt; nxt = tmp;
  }

  // --- H = (1/n) J + P X P  (rank-1 lift back) ---
  iso_rowdot_w<__bf16><<<gRow, blk, 0, stream>>>(cur, t1, w0, w1);
  iso_coldot_w<__bf16><<<gCol, blk, 0, stream>>>(cur, t2, w0, w1);
  iso_wdot_scalar<<<gOne, blk, 0, stream>>>(t1, sscal, w0, w1);
  iso_combine_rank1<__bf16, false, false><<<gRow, blk, 0, stream>>>(
      cur, Xf, t1, t2, sscal, 1.0f / (float)NDIM, nullptr, w0, w1);
}
